// PolicyFullyConnectedGAT_5480378270083
// MI455X (gfx1250) — compile-verified, matrix operand order checked by reading, not run on hardware
//
#include <hip/hip_runtime.h>
#include <hip/hip_fp16.h>

typedef __attribute__((ext_vector_type(16))) _Float16 v16h;
typedef __attribute__((ext_vector_type(8)))  float    v8f;

#define DHID   128
#define NHEAD  8
#define NGRAPH 8
#define FIN    16
#define NT     4   // N-tiles per wave (16x64 strip), independent WMMA accumulators

// ---------- monotonic float<->uint encoding for atomic max ----------
__device__ __forceinline__ unsigned f2ord(float f) {
  unsigned u = __float_as_uint(f);
  return (u & 0x80000000u) ? ~u : (u | 0x80000000u);
}
__device__ __forceinline__ float ord2f(unsigned u) {
  unsigned b = (u & 0x80000000u) ? (u ^ 0x80000000u) : ~u;
  return __uint_as_float(b);
}

// ---------- embedding: leaky(x @ embW + embb, 0.01), writes f32 + f16 ----------
__global__ void k_embed(const float* __restrict__ x, const float* __restrict__ W,
                        const float* __restrict__ b, float* __restrict__ out,
                        _Float16* __restrict__ out16, int Nn) {
  int idx = blockIdx.x * blockDim.x + threadIdx.x;
  if (idx >= Nn * DHID) return;
  int n = idx / DHID, d = idx % DHID;
  float acc = b[d];
#pragma unroll
  for (int f = 0; f < FIN; ++f) acc += x[n * FIN + f] * W[f * DHID + d];
  acc = acc > 0.f ? acc : 0.01f * acc;
  out[idx] = acc;
  out16[idx] = (_Float16)acc;
}

// ---------- weight transpose + f32->f16: Wt[c][k] = W[k][c] ----------
__global__ void k_wt16(const float* __restrict__ W, _Float16* __restrict__ Wt,
                       int K, int Ncols) {
  int idx = blockIdx.x * blockDim.x + threadIdx.x;
  if (idx >= K * Ncols) return;
  int k = idx / Ncols, c = idx % Ncols;
  Wt[(size_t)c * K + k] = (_Float16)W[idx];
}

// ---------- WMMA GEMM: C[MxNcols] = act(A[MxK] @ Bt^T + bias) (+resid) ----------
// A row-major f16, Bt = B^T row-major f16 (Ncols x K). M%16==0, Ncols%64==0, K%32==0.
// One wave per 16x64 strip: A fragment reused across NT=4 independent accumulators
// -> 4 independent v_wmma_f32_16x16x32_f16 per k-step (no serial D->C chain).
__global__ void k_gemm_wmma(const _Float16* __restrict__ A, const _Float16* __restrict__ Bt,
                            const float* __restrict__ bias, const float* __restrict__ resid,
                            float* __restrict__ C, int M, int Ncols, int K, int act) {
  const int lane = threadIdx.x & 31;
  const int wave = threadIdx.x >> 5;
  const int groupsN = Ncols >> 6;  // strips of NT=4 N-tiles
  const int tile = blockIdx.x * (blockDim.x >> 5) + wave;
  const int totalTiles = (M >> 4) * groupsN;
  if (tile >= totalTiles) return;
  const int tm = tile / groupsN, tg = tile % groupsN;
  const int m = (tm << 4) + (lane & 15);
  const int n0 = (tg << 6) + (lane & 15);
  const int kb = (lane >> 4) << 3;  // lanes 0-15: K 0..7/16..23 ; lanes 16-31: K 8..15/24..31
  const _Float16* Ap = A + (size_t)m * K + kb;
  const _Float16* Bp0 = Bt + (size_t)n0 * K + kb;

  v8f zero = {};
  v8f acc[NT];
#pragma unroll
  for (int j = 0; j < NT; ++j) acc[j] = zero;

  for (int k0 = 0; k0 < K; k0 += 32) {
    __builtin_prefetch((const void*)(Ap + 32), 0, 1);  // gfx1250 global_prefetch_b8
    v16h a;
#pragma unroll
    for (int e = 0; e < 8; ++e) { a[e] = Ap[e]; a[e + 8] = Ap[e + 16]; }
    v16h b[NT];
#pragma unroll
    for (int j = 0; j < NT; ++j) {
      const _Float16* Bp = Bp0 + (size_t)(j << 4) * K;
#pragma unroll
      for (int e = 0; e < 8; ++e) { b[j][e] = Bp[e]; b[j][e + 8] = Bp[e + 16]; }
    }
#pragma unroll
    for (int j = 0; j < NT; ++j)
      acc[j] = __builtin_amdgcn_wmma_f32_16x16x32_f16(false, a, false, b[j], (short)0,
                                                      acc[j], false, false);
    Ap += 32;
    Bp0 += 32;
  }

  const int rbase = (tm << 4) + ((lane >> 4) << 3);
#pragma unroll
  for (int j = 0; j < NT; ++j) {
    const int colg = (tg << 6) + (j << 4) + (lane & 15);
    const float bv = bias ? bias[colg] : 0.0f;
#pragma unroll
    for (int i = 0; i < 8; ++i) {
      int row = rbase + i;
      float v = acc[j][i] + bv;
      if (act == 1) v = v > 0.f ? v : 0.01f * v;
      else if (act == 2) v = v > 0.f ? v : 0.f;
      if (resid) v += resid[(size_t)row * Ncols + colg];
      C[(size_t)row * Ncols + colg] = v;
    }
  }
}

// ---------- attention scores: a_s/a_d[n][h] = sum_c Hfeat[n][h][c] * att[h][c] ----------
__global__ void k_att_scores(const float* __restrict__ Hf, const float* __restrict__ atts,
                             const float* __restrict__ attd, float* __restrict__ a_s,
                             float* __restrict__ a_d, int Nn, int H, int C) {
  int idx = blockIdx.x * blockDim.x + threadIdx.x;
  if (idx >= Nn * H) return;
  int n = idx / H, h = idx % H;
  const float* hp = Hf + ((size_t)n * H + h) * C;
  const float* sp = atts + h * C;
  const float* dp = attd + h * C;
  float ss = 0.f, sd = 0.f;
  for (int c = 0; c < C; ++c) { float v = hp[c]; ss += v * sp[c]; sd += v * dp[c]; }
  a_s[idx] = ss;
  a_d[idx] = sd;
}

__global__ void k_init_softmax(unsigned* __restrict__ emax, float* __restrict__ denom,
                               int count) {
  int i = blockIdx.x * blockDim.x + threadIdx.x;
  if (i < count) { emax[i] = f2ord(-3.0e38f); denom[i] = 0.f; }
}

__global__ void k_zero(float* __restrict__ p, size_t count) {
  size_t i = (size_t)blockIdx.x * blockDim.x + threadIdx.x;
  if (i < count) p[i] = 0.f;
}

// ---------- edge phase 1: e = leaky(a_s[s]+a_d[d], 0.2); segment max over dst ----------
__global__ void k_edge_max(const int* __restrict__ src, const int* __restrict__ dst,
                           int E, int Nn, int H, const float* __restrict__ a_s,
                           const float* __restrict__ a_d, float* __restrict__ ebuf,
                           unsigned* __restrict__ emax) {
  int idx = blockIdx.x * blockDim.x + threadIdx.x;
  int T = (E + Nn) * H;
  if (idx >= T) return;
  int e = idx / H, h = idx % H;
  int s, d;
  if (e < E) { s = src[e]; d = dst[e]; } else { s = d = e - E; }  // self loops
  float v = a_s[s * H + h] + a_d[d * H + h];
  v = v > 0.f ? v : 0.2f * v;
  ebuf[idx] = v;
  atomicMax(&emax[d * H + h], f2ord(v));
}

// ---------- edge phase 2: ex = exp(e - max[d]); segment sum denom ----------
__global__ void k_edge_exp(const int* __restrict__ dst, int E, int Nn, int H,
                           float* __restrict__ ebuf, const unsigned* __restrict__ emax,
                           float* __restrict__ denom) {
  int idx = blockIdx.x * blockDim.x + threadIdx.x;
  int T = (E + Nn) * H;
  if (idx >= T) return;
  int e = idx / H, h = idx % H;
  int d = (e < E) ? dst[e] : e - E;
  float ex = expf(ebuf[idx] - ord2f(emax[d * H + h]));
  ebuf[idx] = ex;
  atomicAdd(&denom[d * H + h], ex);
}

// ---------- edge phase 3: out[d][h][:] += Hfeat[s][h][:] * alpha ----------
__global__ void k_edge_aggr(const int* __restrict__ src, const int* __restrict__ dst,
                            int E, int Nn, int H, int C, const float* __restrict__ Hf,
                            const float* __restrict__ ebuf, const float* __restrict__ denom,
                            float* __restrict__ out) {
  int eh = blockIdx.x;
  int e = eh / H, h = eh % H;
  int s, d;
  if (e < E) { s = src[e]; d = dst[e]; } else { s = d = e - E; }
  float alpha = ebuf[eh] / (denom[d * H + h] + 1e-16f);
  const float* hs = Hf + ((size_t)s * H + h) * C;
  float* od = out + ((size_t)d * H + h) * C;
  for (int c = threadIdx.x; c < C; c += blockDim.x) atomicAdd(&od[c], hs[c] * alpha);
}

// ---------- head mean + bias (+resid), optional f16 copy ----------
__global__ void k_head_mean(const float* __restrict__ aggr, const float* __restrict__ bias,
                            const float* __restrict__ resid, float* __restrict__ out,
                            _Float16* __restrict__ out16, int Nn, int H, int C) {
  int idx = blockIdx.x * blockDim.x + threadIdx.x;
  if (idx >= Nn * C) return;
  int n = idx / C, c = idx % C;
  float s = 0.f;
  for (int h = 0; h < H; ++h) s += aggr[((size_t)n * H + h) * C + c];
  s = s / (float)H + bias[c];
  if (resid) s += resid[idx];
  out[idx] = s;
  if (out16) out16[idx] = (_Float16)s;
}

// ---------- BatchNorm stats (one block per column) ----------
__global__ void k_bn_stats(const float* __restrict__ X, int Nn, float* __restrict__ stats) {
  __shared__ float ssum[256], ssq[256];
  int d = blockIdx.x;
  float sum = 0.f, sq = 0.f;
  for (int n = threadIdx.x; n < Nn; n += blockDim.x) {
    float v = X[(size_t)n * DHID + d];
    sum += v; sq += v * v;
  }
  ssum[threadIdx.x] = sum; ssq[threadIdx.x] = sq;
  __syncthreads();
  for (int s = blockDim.x >> 1; s > 0; s >>= 1) {
    if (threadIdx.x < (unsigned)s) {
      ssum[threadIdx.x] += ssum[threadIdx.x + s];
      ssq[threadIdx.x]  += ssq[threadIdx.x + s];
    }
    __syncthreads();
  }
  if (threadIdx.x == 0) {
    float mu = ssum[0] / (float)Nn;
    float var = ssq[0] / (float)Nn - mu * mu;
    stats[d] = mu;
    stats[DHID + d] = rsqrtf(var + 1e-5f);
  }
}

__global__ void k_bn_apply(const float* __restrict__ X, const float* __restrict__ stats,
                           const float* __restrict__ g, const float* __restrict__ be,
                           float* __restrict__ out, _Float16* __restrict__ out16, int Nn) {
  int idx = blockIdx.x * blockDim.x + threadIdx.x;
  if (idx >= Nn * DHID) return;
  int d = idx % DHID;
  float y = (X[idx] - stats[d]) * stats[DHID + d] * g[d] + be[d];
  out[idx] = y;
  if (out16) out16[idx] = (_Float16)y;
}

// ---------- decoder projection: hf = x3 @ decW ; a_s = hf*decas ; a_d = hf*decad ----------
__global__ void k_dec_proj(const float* __restrict__ X, const float* __restrict__ W,
                           const float* __restrict__ as_, const float* __restrict__ ad_,
                           float* __restrict__ hf, float* __restrict__ a_s,
                           float* __restrict__ a_d, int Nn) {
  int n = blockIdx.x * blockDim.x + threadIdx.x;
  if (n >= Nn) return;
  float acc = 0.f;
  for (int d = 0; d < DHID; ++d) acc += X[(size_t)n * DHID + d] * W[d];
  hf[n] = acc;
  a_s[n] = acc * as_[0];
  a_d[n] = acc * ad_[0];
}

// ---------- graph pooling (segment sum + counts) ----------
__global__ void k_pool(const float* __restrict__ X, const int* __restrict__ batch,
                       float* __restrict__ pooled, float* __restrict__ counts, int Nn) {
  int idx = blockIdx.x * blockDim.x + threadIdx.x;
  if (idx >= Nn * DHID) return;
  int n = idx / DHID, d = idx % DHID;
  int g = batch[n];
  atomicAdd(&pooled[(size_t)g * DHID + d], X[idx]);
  if (d == 0) atomicAdd(&counts[g], 1.0f);
}

// ---------- value head: relu(mean_pool @ vW1 + vb1) @ vW2 + vb2 ----------
__global__ void k_value(const float* __restrict__ pooled, const float* __restrict__ counts,
                        const float* __restrict__ W1, const float* __restrict__ b1,
                        const float* __restrict__ W2, const float* __restrict__ b2,
                        float* __restrict__ out) {
  __shared__ float red[128];
  int g = blockIdx.x;
  int j = threadIdx.x;  // 128 threads
  float inv = 1.0f / counts[g];
  float acc = b1[j];
  for (int d = 0; d < DHID; ++d) acc += pooled[(size_t)g * DHID + d] * inv * W1[d * 128 + j];
  acc = acc > 0.f ? acc : 0.f;
  red[j] = acc * W2[j];
  __syncthreads();
  for (int s = 64; s > 0; s >>= 1) {
    if (j < s) red[j] += red[j + s];
    __syncthreads();
  }
  if (j == 0) out[g] = red[0] + b2[0];
}

extern "C" void kernel_launch(void* const* d_in, const int* in_sizes, int n_in,
                              void* d_out, int out_size, void* d_ws, size_t ws_size,
                              hipStream_t stream) {
  (void)n_in; (void)out_size; (void)ws_size;
  const float* x     = (const float*)d_in[0];
  const int*   ei    = (const int*)d_in[1];
  const int*   batch = (const int*)d_in[2];
  const float* embW  = (const float*)d_in[3];
  const float* embb  = (const float*)d_in[4];
  const float* decW  = (const float*)d_in[29];
  const float* decas = (const float*)d_in[30];
  const float* decad = (const float*)d_in[31];
  const float* decb  = (const float*)d_in[32];
  const float* vW1   = (const float*)d_in[33];
  const float* vb1   = (const float*)d_in[34];
  const float* vW2   = (const float*)d_in[35];
  const float* vb2   = (const float*)d_in[36];

  const int Nn = in_sizes[0] / FIN;
  const int E  = in_sizes[1] / 2;
  const int* src = ei;
  const int* dst = ei + E;
  float* dout = (float*)d_out;

  // ---- workspace carve (256B aligned) ----
  char* base = (char*)d_ws;
  size_t off = 0;
  auto carve = [&](size_t bytes) -> char* {
    char* p = base + off;
    off += (bytes + 255) & ~(size_t)255;
    return p;
  };
  const size_t ND = (size_t)Nn * DHID;
  float*    P0   = (float*)carve(ND * 4);
  float*    P1   = (float*)carve(ND * 4);
  float*    P2   = (float*)carve(ND * 4);
  _Float16* P0h  = (_Float16*)carve(ND * 2);
  _Float16* P1h  = (_Float16*)carve(ND * 2);
  _Float16* P2h  = (_Float16*)carve(ND * 2);
  float*    Hfeat = (float*)carve(ND * NHEAD * 4);
  float*    aggr  = (float*)carve(ND * NHEAD * 4);
  float*    a_s   = (float*)carve((size_t)Nn * NHEAD * 4);
  float*    a_d   = (float*)carve((size_t)Nn * NHEAD * 4);
  unsigned* emax  = (unsigned*)carve((size_t)Nn * NHEAD * 4);
  float*    denom = (float*)carve((size_t)Nn * NHEAD * 4);
  float*    ebuf  = (float*)carve((size_t)(E + Nn) * NHEAD * 4);
  _Float16* Wt    = (_Float16*)carve((size_t)DHID * NHEAD * DHID * 2);
  float*    stats = (float*)carve(DHID * 2 * 4);
  float*    pooled = (float*)carve((size_t)NGRAPH * DHID * 4);
  float*    counts = (float*)carve(NGRAPH * 4);
  float*    hfdec  = (float*)carve((size_t)Nn * 4);

  auto run_gat = [&](const _Float16* in16, const float* W, const float* atts,
                     const float* attd, const float* bias, const float* resid,
                     float* out, _Float16* out16) {
    const int HD = NHEAD * DHID;
    int tot = DHID * HD;
    k_wt16<<<(tot + 255) / 256, 256, 0, stream>>>(W, Wt, DHID, HD);
    int tiles = (Nn / 16) * (HD / 64);  // 16x64 strips per wave
    k_gemm_wmma<<<(tiles + 7) / 8, 256, 0, stream>>>(in16, Wt, nullptr, nullptr, Hfeat,
                                                     Nn, HD, DHID, 0);
    int nh = Nn * NHEAD;
    k_att_scores<<<(nh + 255) / 256, 256, 0, stream>>>(Hfeat, atts, attd, a_s, a_d,
                                                       Nn, NHEAD, DHID);
    k_init_softmax<<<(nh + 255) / 256, 256, 0, stream>>>(emax, denom, nh);
    size_t an = ND * NHEAD;
    k_zero<<<(unsigned)((an + 255) / 256), 256, 0, stream>>>(aggr, an);
    int T = (E + Nn) * NHEAD;
    k_edge_max<<<(T + 255) / 256, 256, 0, stream>>>(src, dst, E, Nn, NHEAD, a_s, a_d,
                                                    ebuf, emax);
    k_edge_exp<<<(T + 255) / 256, 256, 0, stream>>>(dst, E, Nn, NHEAD, ebuf, emax, denom);
    k_edge_aggr<<<T, 128, 0, stream>>>(src, dst, E, Nn, NHEAD, DHID, Hfeat, ebuf, denom,
                                       aggr);
    k_head_mean<<<(int)((ND + 255) / 256), 256, 0, stream>>>(aggr, bias, resid, out,
                                                             out16, Nn, NHEAD, DHID);
  };

  auto run_ff = [&](const _Float16* in16, const float* fW, const float* fb,
                    const float* resid, float* out, int act) {
    k_wt16<<<(DHID * DHID + 255) / 256, 256, 0, stream>>>(fW, Wt, DHID, DHID);
    int tiles = (Nn / 16) * (DHID / 64);
    k_gemm_wmma<<<(tiles + 7) / 8, 256, 0, stream>>>(in16, Wt, fb, resid, out,
                                                     Nn, DHID, DHID, act);
  };

  auto run_bn = [&](float* X, const float* g, const float* be, _Float16* out16) {
    k_bn_stats<<<DHID, 256, 0, stream>>>(X, Nn, stats);
    k_bn_apply<<<(int)((ND + 255) / 256), 256, 0, stream>>>(X, stats, g, be, X, out16, Nn);
  };

  // layer param index helpers: layer l in {0,1,2} starts at 5 + 8*l
  auto LP = [&](int l, int j) { return (const float*)d_in[5 + 8 * l + j]; };
  // j: 0=W 1=as 2=ad 3=b 4=fW 5=fb 6=g 7=be

  // ---- forward ----
  k_embed<<<(int)((ND + 255) / 256), 256, 0, stream>>>(x, embW, embb, P0, P0h, Nn);

  // block 1 (encoder1 applied twice, LeakyReLU ff)
  run_gat(P0h, LP(0, 0), LP(0, 1), LP(0, 2), LP(0, 3), nullptr, P1, P1h);   // h1
  run_gat(P1h, LP(0, 0), LP(0, 1), LP(0, 2), LP(0, 3), P1, P2, P2h);        // t = gat(h1)+h1
  run_ff(P2h, LP(0, 4), LP(0, 5), P1, P0, /*leaky*/ 1);                     // + h1
  run_bn(P0, LP(0, 6), LP(0, 7), P0h);                                      // x1 in P0

  // block 2 (ReLU ff)
  run_gat(P0h, LP(1, 0), LP(1, 1), LP(1, 2), LP(1, 3), P0, P1, P1h);        // t2
  run_ff(P1h, LP(1, 4), LP(1, 5), P0, P2, /*relu*/ 2);
  run_bn(P2, LP(1, 6), LP(1, 7), P2h);                                      // x2 in P2

  // block 3 (ReLU ff)
  run_gat(P2h, LP(2, 0), LP(2, 1), LP(2, 2), LP(2, 3), P2, P1, P1h);        // t3
  run_ff(P1h, LP(2, 4), LP(2, 5), P2, P0, /*relu*/ 2);
  run_bn(P0, LP(2, 6), LP(2, 7), nullptr);                                  // x3 in P0

  // decoder GAT (H=1, C=1) -> dout[0..N)
  k_dec_proj<<<(Nn + 255) / 256, 256, 0, stream>>>(P0, decW, decas, decad, hfdec,
                                                   a_s, a_d, Nn);
  k_init_softmax<<<(Nn + 255) / 256, 256, 0, stream>>>(emax, denom, Nn);
  k_zero<<<(Nn + 255) / 256, 256, 0, stream>>>(aggr, (size_t)Nn);
  int T1 = E + Nn;
  k_edge_max<<<(T1 + 255) / 256, 256, 0, stream>>>(src, dst, E, Nn, 1, a_s, a_d, ebuf,
                                                   emax);
  k_edge_exp<<<(T1 + 255) / 256, 256, 0, stream>>>(dst, E, Nn, 1, ebuf, emax, denom);
  k_edge_aggr<<<T1, 32, 0, stream>>>(src, dst, E, Nn, 1, 1, hfdec, ebuf, denom, aggr);
  k_head_mean<<<(Nn + 255) / 256, 256, 0, stream>>>(aggr, decb, nullptr, dout, nullptr,
                                                    Nn, 1, 1);

  // value head -> dout[N..N+G)
  k_zero<<<((NGRAPH * DHID) + 255) / 256, 256, 0, stream>>>(pooled,
                                                            (size_t)NGRAPH * DHID);
  k_zero<<<1, 256, 0, stream>>>(counts, (size_t)NGRAPH);
  k_pool<<<(int)((ND + 255) / 256), 256, 0, stream>>>(P0, batch, pooled, counts, Nn);
  k_value<<<NGRAPH, 128, 0, stream>>>(pooled, counts, vW1, vb1, vW2, vb2, dout + Nn);
}
